// GAT_11450382811259
// MI455X (gfx1250) — matrix operand order repaired and checked by reading, not compiled, not run on hardware
//
#include <hip/hip_runtime.h>
#include <hip/hip_fp16.h>
#include <math.h>

// ---- problem constants (from reference) ----
#define NN      8192        // N nodes
#define F_IN    256
#define HID     64
#define HEADS   8
#define C_TOT   (HEADS*HID) // 512
#define BN_EPS  1e-3f
#define ALPHA   0.2f        // leaky relu slope
#define KSPLIT  4           // waves per row-tile (split over m)
#define MSEG    (NN / KSPLIT)

typedef __attribute__((ext_vector_type(16))) _Float16 v16h;
typedef __attribute__((ext_vector_type(8)))  _Float16 v8h;
typedef __attribute__((ext_vector_type(8)))  float    v8f;

// ---------------------------------------------------------------------------
// Kernel 0: fold BatchNorm (inference) into per-channel scale/shift
// ---------------------------------------------------------------------------
__global__ __launch_bounds__(512)
void bn_prep_kernel(const float* __restrict__ gamma, const float* __restrict__ beta,
                    const float* __restrict__ mmean, const float* __restrict__ mvar,
                    float* __restrict__ bnS, float* __restrict__ bnB)
{
    int c = threadIdx.x;
    float s = gamma[c] * rsqrtf(mvar[c] + BN_EPS);
    bnS[c] = s;
    bnB[c] = beta[c] - mmean[c] * s;
}

// ---------------------------------------------------------------------------
// Kernel 1: seq_fts = inputs @ W1 + b1 (WMMA f16->f32), stored transposed as
// f16 seqT[h][n]; also f[n] = seq_fts[n,:] @ W2 + b2.
// One wave per 16-row tile. grid = 512 x 32 threads.
// ---------------------------------------------------------------------------
__global__ __launch_bounds__(32)
void gemm1_kernel(const float* __restrict__ inp, const float* __restrict__ W1,
                  const float* __restrict__ b1,  const float* __restrict__ W2,
                  const float* __restrict__ b2,
                  _Float16* __restrict__ seqT, float* __restrict__ fvec)
{
    __shared__ float lds[16 * 33];            // staged input tile, padded (conflict-free)
    const int lane = threadIdx.x;             // 0..31 (wave32)
    const int g    = lane >> 4;               // lane group 0/1
    const int l15  = lane & 15;
    const int n0   = blockIdx.x * 16;

    v8f acc[4] = {v8f{}, v8f{}, v8f{}, v8f{}};

    for (int kc = 0; kc < F_IN; kc += 32) {
        // ---- stage 16x32 fp32 input tile (coalesced rows) ----
        #pragma unroll
        for (int r = 0; r < 16; ++r)
            lds[r * 33 + lane] = inp[(size_t)(n0 + r) * F_IN + kc + lane];
        __syncthreads();

        // ---- WMMA A operand (16x32 f16): lane=row l15, K = e+16g
        v16h a;
        #pragma unroll
        for (int e = 0; e < 16; ++e) {
            int k = e + 16 * g;
            a[e] = (_Float16)lds[l15 * 33 + k];
        }

        // ---- 4 B tiles of W1 chunk (32x16 each), K = e + 16g, col h = 16t+l15
        #pragma unroll
        for (int t = 0; t < 4; ++t) {
            v16h b;
            #pragma unroll
            for (int e = 0; e < 16; ++e)
                b[e] = (_Float16)W1[(size_t)(kc + e + 16 * g) * HID + 16 * t + l15];
            acc[t] = __builtin_amdgcn_wmma_f32_16x16x32_f16(
                false, a, false, b, (short)0, acc[t], false, false);
        }
        __syncthreads();
    }

    // ---- epilogue: +b1, store seqT (f16, transposed), compute f = seq@W2+b2
    float b1v[4], w2v[4];
    #pragma unroll
    for (int t = 0; t < 4; ++t) { b1v[t] = b1[16 * t + l15]; w2v[t] = W2[16 * t + l15]; }

    float fpart[8] = {0.f,0.f,0.f,0.f,0.f,0.f,0.f,0.f};
    #pragma unroll
    for (int t = 0; t < 4; ++t) {
        const int h = 16 * t + l15;
        #pragma unroll
        for (int r = 0; r < 8; ++r) {
            float v = acc[t][r] + b1v[t];             // C layout: row = r+8g, col = l15
            const int n = n0 + r + 8 * g;
            seqT[(size_t)h * NN + n] = (_Float16)v;
            fpart[r] += v * w2v[t];
        }
    }
    #pragma unroll
    for (int mask = 1; mask <= 8; mask <<= 1) {
        #pragma unroll
        for (int r = 0; r < 8; ++r)
            fpart[r] += __shfl_xor(fpart[r], mask, 32);
    }
    if (l15 == 0) {
        float b2v = b2[0];
        #pragma unroll
        for (int r = 0; r < 8; ++r)
            fvec[n0 + r + 8 * g] = fpart[r] + b2v;
    }
}

// ---------------------------------------------------------------------------
// Kernel 2: split-K fused flash attention. 4 waves per 16-row tile, each wave
// covers m-segment of 2048 with its own online softmax, then LDS merge.
// grid = 512 x 128 threads.
// ---------------------------------------------------------------------------
__global__ __launch_bounds__(128)
void attn_kernel(const float* __restrict__ bias, const _Float16* __restrict__ seqT,
                 const float* __restrict__ fvec, const float* __restrict__ bnS,
                 const float* __restrict__ bnB,  float* __restrict__ out)
{
    __shared__ float s_bias[KSPLIT * 16 * 33];   // per-wave staged bias tile
    __shared__ float s_f[KSPLIT * 32];           // per-wave staged f chunk
    __shared__ float s_bnS[C_TOT];
    __shared__ float s_bnB[C_TOT];
    __shared__ float s_M[KSPLIT * 16];           // per-wave row max
    __shared__ float s_l[KSPLIT * 16];           // per-wave row sum
    __shared__ float s_acc[KSPLIT * 32 * 33];    // per-wave rescaled accumulators (padded)

    const int tid  = threadIdx.x;
    const int wave = tid >> 5;                   // 0..3
    const int lane = tid & 31;
    const int g    = lane >> 4;
    const int l15  = lane & 15;
    const int n0   = blockIdx.x * 16;

    for (int i = tid; i < C_TOT; i += 128) { s_bnS[i] = bnS[i]; s_bnB[i] = bnB[i]; }

    float* my_bias = s_bias + wave * (16 * 33);
    float* my_f    = s_f    + wave * 32;

    const float fn = fvec[n0 + l15];             // f for this lane's row
    float M = -INFINITY;                         // running row max (row = l15)
    float l = 0.f;                               // running row sum
    v8f acc[4] = {v8f{}, v8f{}, v8f{}, v8f{}};

    const int mbase0 = wave * MSEG;
    for (int base = mbase0; base < mbase0 + MSEG; base += 32) {
        // stage f chunk + 16x32 bias tile (coalesced, padded LDS)
        my_f[lane] = fvec[base + lane];
        #pragma unroll
        for (int r = 0; r < 16; ++r)
            my_bias[r * 33 + lane] = bias[(size_t)(n0 + r) * NN + base + lane];
        __syncthreads();

        // scores in WMMA-A register layout; per-lane partial max
        float s[16];
        float tmax = -INFINITY;
        #pragma unroll
        for (int e = 0; e < 16; ++e) {
            int k = e + (e < 8 ? 0 : 8) + 8 * g;          // A layout K index
            float logit = fn + my_f[k];
            float sc = (logit > 0.f ? logit : ALPHA * logit) + my_bias[l15 * 33 + k];
            s[e] = sc;
            tmax = fmaxf(tmax, sc);
        }
        tmax = fmaxf(tmax, __shfl_xor(tmax, 16, 32));     // join the two K-halves
        float Mnew  = fmaxf(M, tmax);
        float scale = __expf(M - Mnew);                   // 0 on first chunk

        float tsum = 0.f;
        v16h a;
        #pragma unroll
        for (int e = 0; e < 16; ++e) {
            float p = __expf(s[e] - Mnew);
            tsum += p;
            a[e] = (_Float16)p;                           // P in A layout, f16
        }
        tsum += __shfl_xor(tsum, 16, 32);
        l = l * scale + tsum;
        M = Mnew;

        // rescale accumulators: acc VGPR r holds row r+8g -> scale at lane r+8g
        float sc_r[8];
        #pragma unroll
        for (int r = 0; r < 8; ++r) sc_r[r] = __shfl(scale, r + 8 * g, 32);
        #pragma unroll
        for (int t = 0; t < 4; ++t) {
            #pragma unroll
            for (int r = 0; r < 8; ++r) acc[t][r] *= sc_r[r];
        }

        // B operands from transposed f16 seq_fts: col h=16t+l15, K=e+(e<8?0:8)+8g
        #pragma unroll
        for (int t = 0; t < 4; ++t) {
            const _Float16* vp = seqT + (size_t)(16 * t + l15) * NN + base + 8 * g;
            v8h blo = *(const v8h*)(vp);
            v8h bhi = *(const v8h*)(vp + 16);
            v16h b;
            #pragma unroll
            for (int e = 0; e < 8; ++e) { b[e] = blo[e]; b[e + 8] = bhi[e]; }
            acc[t] = __builtin_amdgcn_wmma_f32_16x16x32_f16(
                false, a, false, b, (short)0, acc[t], false, false);
        }
        __syncthreads();
    }

    // ---- split-K merge across the 4 waves ----
    if (lane < 16) { s_M[wave * 16 + lane] = M; s_l[wave * 16 + lane] = l; }
    __syncthreads();

    float Mg = s_M[l15];
    #pragma unroll
    for (int w = 1; w < KSPLIT; ++w) Mg = fmaxf(Mg, s_M[w * 16 + l15]);
    float myscale = __expf(M - Mg);

    float sc_r[8];
    #pragma unroll
    for (int r = 0; r < 8; ++r) sc_r[r] = __shfl(myscale, r + 8 * g, 32);
    #pragma unroll
    for (int t = 0; t < 4; ++t) {
        #pragma unroll
        for (int r = 0; r < 8; ++r)
            s_acc[wave * (32 * 33) + lane * 33 + t * 8 + r] = acc[t][r] * sc_r[r];
    }
    __syncthreads();

    float lg = 0.f;
    #pragma unroll
    for (int w = 0; w < KSPLIT; ++w)
        lg += s_l[w * 16 + l15] * __expf(s_M[w * 16 + l15] - Mg);
    float linv = 1.0f / lg;
    float il[8];
    #pragma unroll
    for (int r = 0; r < 8; ++r) il[r] = __shfl(linv, r + 8 * g, 32);

    // each wave finalizes one 16-wide h-tile: t = wave
    {
        const int t = wave;
        const int h = 16 * t + l15;
        #pragma unroll
        for (int r = 0; r < 8; ++r) {
            float v = 0.f;
            #pragma unroll
            for (int w = 0; w < KSPLIT; ++w)
                v += s_acc[w * (32 * 33) + lane * 33 + t * 8 + r];
            v *= il[r];
            float ho = v > 0.f ? v : (__expf(v) - 1.f);   // elu
            const int n = n0 + r + 8 * g;
            #pragma unroll
            for (int hd = 0; hd < HEADS; ++hd) {
                const int c = hd * HID + h;
                out[(size_t)n * C_TOT + c] = ho * s_bnS[c] + s_bnB[c];
            }
        }
    }
}

// ---------------------------------------------------------------------------
extern "C" void kernel_launch(void* const* d_in, const int* in_sizes, int n_in,
                              void* d_out, int out_size, void* d_ws, size_t ws_size,
                              hipStream_t stream)
{
    const float* inputs = (const float*)d_in[0];   // [1,8192,256]
    const float* bias   = (const float*)d_in[1];   // [1,8192,8192]
    const float* W1     = (const float*)d_in[2];   // [256,64]
    const float* b1     = (const float*)d_in[3];   // [64]
    const float* W2     = (const float*)d_in[4];   // [64,1]
    const float* b2     = (const float*)d_in[5];   // [1]
    const float* gamma  = (const float*)d_in[6];   // [512]
    const float* beta   = (const float*)d_in[7];
    const float* mmean  = (const float*)d_in[8];
    const float* mvar   = (const float*)d_in[9];
    float* out = (float*)d_out;                    // [1,8192,512] fp32

    // workspace carve: f (32KB) | bnS/bnB (4KB) | seqT f16 (1MB)
    char* ws = (char*)d_ws;
    float*     fvec = (float*)ws;                        // 8192 f32
    float*     bnS  = (float*)(ws + 32 * 1024);          // 512 f32
    float*     bnB  = bnS + C_TOT;                       // 512 f32
    _Float16*  seqT = (_Float16*)(ws + 40 * 1024);       // 64*8192 f16

    bn_prep_kernel<<<1, C_TOT, 0, stream>>>(gamma, beta, mmean, mvar, bnS, bnB);
    gemm1_kernel<<<NN / 16, 32, 0, stream>>>(inputs, W1, b1, W2, b2, seqT, fvec);
    attn_kernel<<<NN / 16, 128, 0, stream>>>(bias, seqT, fvec, bnS, bnB, out);
}
